// MultiScaleRetention_67276367725256
// MI455X (gfx1250) — matrix operand order repaired and checked by reading, not compiled, not run on hardware
//
#include <hip/hip_runtime.h>
#include <hip/hip_bf16.h>
#include <cstdint>
#include <cstddef>

// ---------------------------------------------------------------------------
// MultiScaleRetention on MI455X (gfx1250), wave32 + WMMA bf16 path,
// async global->LDS double-buffered GEMM pipeline.
// ---------------------------------------------------------------------------

typedef __bf16 v16bf __attribute__((ext_vector_type(16)));
typedef __bf16 v8bf  __attribute__((ext_vector_type(8)));
typedef __bf16 v4bf  __attribute__((ext_vector_type(4)));
typedef float  v8f   __attribute__((ext_vector_type(8)));

#define D_MODEL  2048
#define NHEAD    16
#define HEAD_DIM 128
#define BATCH    4
#define SEQ      4096
#define MROWS    (BATCH * SEQ)   // 16384

#define BM   128
#define BN   128
#define BK   32
#define LDSS 40                  // padded LDS row stride in bf16 (20 dwords)

// Async copy: 16B per lane, global -> LDS, no VGPR round trip (ASYNCcnt).
__device__ __forceinline__ void async_ld_b128(unsigned int lds_addr,
                                              const void* gptr) {
    asm volatile("global_load_async_to_lds_b128 %0, %1, off"
                 :: "v"(lds_addr), "v"(gptr) : "memory");
}
// Low 32 bits of a flat shared pointer == LDS byte address (ISA 10.2 aperture).
__device__ __forceinline__ unsigned int lds_addr_of(const void* p) {
    return (unsigned int)(uintptr_t)p;
}

// ---------------------------------------------------------------------------
// fp32 -> bf16 convert (vectorized: float4 in, 4x bf16 out)
// ---------------------------------------------------------------------------
__global__ void cvt_f32_bf16(const float* __restrict__ in,
                             __bf16* __restrict__ out, long n4) {
    long i = (long)blockIdx.x * blockDim.x + threadIdx.x;
    if (i < n4) {
        float4 f = ((const float4*)in)[i];
        v4bf v;
        v[0] = (__bf16)f.x; v[1] = (__bf16)f.y;
        v[2] = (__bf16)f.z; v[3] = (__bf16)f.w;
        ((v4bf*)out)[i] = v;
    }
}

// ---------------------------------------------------------------------------
// Tiled bf16 WMMA GEMM: C[M,N] = A[M,K] * W[N,K]^T   (y = x @ W.T)
// mode 0: store fp32; mode 1: store silu(fp32)
// 256 threads = 8 waves; 128x128 block tile; each wave owns 32x64 = 2x4
// accumulators of v_wmma_f32_16x16x32_bf16. Double-buffered LDS filled by
// global_load_async_to_lds_b128, synchronized with s_wait_asynccnt.
// ---------------------------------------------------------------------------
__global__ __launch_bounds__(256)
void gemm_bf16_wmma(const __bf16* __restrict__ A,
                    const __bf16* __restrict__ W,
                    float* __restrict__ C,
                    int M, int N, int K, int mode) {
    __shared__ __align__(16) __bf16 sA[2][BM * LDSS];
    __shared__ __align__(16) __bf16 sB[2][BN * LDSS];

    const int tid  = threadIdx.x;
    const int lane = tid & 31;
    const int wv   = tid >> 5;     // wave id 0..7
    const int wm   = wv >> 1;      // 0..3 : 32-row strip
    const int wn   = wv & 1;       // 0..1 : 64-col strip
    const int l15  = lane & 15;
    const int hsel = lane >> 4;    // 0 / 1 (lane half)

    const int bM = blockIdx.x * BM;
    const int bN = blockIdx.y * BN;

    // ---- per-thread staging coordinates (two 16B chunks per tile per array)
    const int i0 = tid * 8,          r0 = i0 >> 5, c0 = i0 & 31;
    const int i1 = (256 + tid) * 8,  r1 = i1 >> 5, c1 = i1 & 31;
    const __bf16* gA0 = A + (size_t)(bM + r0) * K + c0;
    const __bf16* gA1 = A + (size_t)(bM + r1) * K + c1;
    const __bf16* gB0 = W + (size_t)(bN + r0) * K + c0;
    const __bf16* gB1 = W + (size_t)(bN + r1) * K + c1;
    unsigned int lA0[2], lA1[2], lB0[2], lB1[2];
#pragma unroll
    for (int b = 0; b < 2; ++b) {
        lA0[b] = lds_addr_of(&sA[b][r0 * LDSS + c0]);
        lA1[b] = lds_addr_of(&sA[b][r1 * LDSS + c1]);
        lB0[b] = lds_addr_of(&sB[b][r0 * LDSS + c0]);
        lB1[b] = lds_addr_of(&sB[b][r1 * LDSS + c1]);
    }

    v8f acc[2][4];
#pragma unroll
    for (int i = 0; i < 2; ++i)
#pragma unroll
        for (int j = 0; j < 4; ++j)
#pragma unroll
            for (int r = 0; r < 8; ++r) acc[i][j][r] = 0.0f;

    const int nk = K / BK;
    // prologue: fill buffer 0 (4 async instructions per wave)
    async_ld_b128(lA0[0], gA0);
    async_ld_b128(lA1[0], gA1);
    async_ld_b128(lB0[0], gB0);
    async_ld_b128(lB1[0], gB1);

    for (int kt = 0; kt < nk; ++kt) {
        const int buf = kt & 1;
        if (kt + 1 < nk) {
            const int k1 = (kt + 1) * BK;
            async_ld_b128(lA0[buf ^ 1], gA0 + k1);
            async_ld_b128(lA1[buf ^ 1], gA1 + k1);
            async_ld_b128(lB0[buf ^ 1], gB0 + k1);
            async_ld_b128(lB1[buf ^ 1], gB1 + k1);
            // <=4 outstanding: current tile's copies complete, next still flying
            asm volatile("s_wait_asynccnt 0x4" ::: "memory");
        } else {
            asm volatile("s_wait_asynccnt 0x0" ::: "memory");
        }
        __syncthreads();   // all waves' copies for `buf` visible in LDS

        // ---- fragment assembly per documented CDNA5 16-bit layouts
        v16bf af[2];
#pragma unroll
        for (int mt = 0; mt < 2; ++mt) {
            const __bf16* pa = &sA[buf][(wm * 32 + mt * 16 + l15) * LDSS + hsel * 8];
            v8bf lo = *(const v8bf*)(pa);
            v8bf hi = *(const v8bf*)(pa + 16);
            af[mt] = __builtin_shufflevector(lo, hi,
                      0,1,2,3,4,5,6,7,8,9,10,11,12,13,14,15);
        }
        v16bf bfg[4];
#pragma unroll
        for (int nt = 0; nt < 4; ++nt) {
            const __bf16* pb = &sB[buf][(wn * 64 + nt * 16 + l15) * LDSS + hsel * 8];
            v8bf lo = *(const v8bf*)(pb);
            v8bf hi = *(const v8bf*)(pb + 16);
            bfg[nt] = __builtin_shufflevector(lo, hi,
                       0,1,2,3,4,5,6,7,8,9,10,11,12,13,14,15);
        }

        // ---- matrix core: D = A*B + C  (f32 accumulate)
#pragma unroll
        for (int mt = 0; mt < 2; ++mt)
#pragma unroll
            for (int nt = 0; nt < 4; ++nt)
                acc[mt][nt] = __builtin_amdgcn_wmma_f32_16x16x32_bf16(
                    false, af[mt], false, bfg[nt],
                    (short)0, acc[mt][nt], false, false);
        __syncthreads();   // done reading `buf` before it is refilled
    }

    // ---- epilogue: documented C/D layout  M = r + 8*hsel, N = l15
#pragma unroll
    for (int mt = 0; mt < 2; ++mt)
#pragma unroll
        for (int nt = 0; nt < 4; ++nt)
#pragma unroll
            for (int r = 0; r < 8; ++r) {
                int row = bM + wm * 32 + mt * 16 + hsel * 8 + r;
                int col = bN + wn * 64 + nt * 16 + l15;
                float v = acc[mt][nt][r];
                if (mode == 1) v = v / (1.0f + __expf(-v));   // SiLU
                C[(size_t)row * N + col] = v;
            }
}

// ---------------------------------------------------------------------------
// In-place retention scan: state = lam_h * state + v_t  along L.
// One thread per (batch, channel); coalesced 128B per wave per step.
// ---------------------------------------------------------------------------
__global__ void retention_scan(float* __restrict__ v,
                               const float* __restrict__ beta) {
    int c  = blockIdx.x * blockDim.x + threadIdx.x;    // 0 .. BATCH*D_MODEL-1
    int b  = c / D_MODEL;
    int ch = c % D_MODEL;
    int h  = ch / HEAD_DIM;
    float lam = 1.0f - exp2f(-beta[h]);
    lam = fminf(fmaxf(lam, 1.175494351e-38f), 1.0f - 1e-9f);

    float state = 0.0f;
    float* p = v + (size_t)b * SEQ * D_MODEL + ch;
    for (int t = 0; t < SEQ; t += 4) {               // batch 4 loads ahead of
        float a0 = p[0 * D_MODEL];                   // the serial FMA chain
        float a1 = p[1 * D_MODEL];
        float a2 = p[2 * D_MODEL];
        float a3 = p[3 * D_MODEL];
        state = lam * state + a0; p[0 * D_MODEL] = state;
        state = lam * state + a1; p[1 * D_MODEL] = state;
        state = lam * state + a2; p[2 * D_MODEL] = state;
        state = lam * state + a3; p[3 * D_MODEL] = state;
        p += 4 * D_MODEL;
    }
}

// ---------------------------------------------------------------------------
// Fused: y = LayerNorm(q * ret) * gamma + beta_ln, then y *= silu_gate; -> bf16
// ---------------------------------------------------------------------------
__global__ __launch_bounds__(256)
void ln_gate_kernel(const float* __restrict__ q,
                    const float* __restrict__ ret,
                    const float* __restrict__ g,
                    const float* __restrict__ gamma,
                    const float* __restrict__ betaln,
                    __bf16* __restrict__ yb) {
    const int    row  = blockIdx.x;
    const size_t base = (size_t)row * D_MODEL;

    float y[8];
    float s = 0.0f, s2 = 0.0f;
#pragma unroll
    for (int i = 0; i < 8; ++i) {
        int idx = threadIdx.x + i * 256;
        float v = q[base + idx] * ret[base + idx];
        y[i] = v;
        s += v;
        s2 += v * v;
    }
#pragma unroll
    for (int off = 16; off > 0; off >>= 1) {   // wave32 reduce
        s  += __shfl_xor(s,  off, 32);
        s2 += __shfl_xor(s2, off, 32);
    }
    __shared__ float rs[8], rq[8];
    int wv = threadIdx.x >> 5, lane = threadIdx.x & 31;
    if (lane == 0) { rs[wv] = s; rq[wv] = s2; }
    __syncthreads();
    float ts = 0.0f, tq = 0.0f;
#pragma unroll
    for (int i = 0; i < 8; ++i) { ts += rs[i]; tq += rq[i]; }

    const float mu   = ts * (1.0f / D_MODEL);
    const float var  = tq * (1.0f / D_MODEL) - mu * mu;
    const float rstd = rsqrtf(var + 1e-5f);
#pragma unroll
    for (int i = 0; i < 8; ++i) {
        int idx = threadIdx.x + i * 256;
        float o = (y[i] - mu) * rstd * gamma[idx] + betaln[idx];
        o *= g[base + idx];                    // gate already silu'd in GEMM epilogue
        yb[base + idx] = (__bf16)o;
    }
}

// ---------------------------------------------------------------------------
// Host launcher. Inputs (dict order): x, Wq, Wv, Wo, Wg, ln_gamma, ln_beta, beta
// ---------------------------------------------------------------------------
extern "C" void kernel_launch(void* const* d_in, const int* in_sizes, int n_in,
                              void* d_out, int out_size, void* d_ws, size_t ws_size,
                              hipStream_t stream) {
    const float* x    = (const float*)d_in[0];
    const float* Wq   = (const float*)d_in[1];
    const float* Wv   = (const float*)d_in[2];
    const float* Wo   = (const float*)d_in[3];
    const float* Wg   = (const float*)d_in[4];
    const float* gma  = (const float*)d_in[5];
    const float* btl  = (const float*)d_in[6];
    const float* beta = (const float*)d_in[7];
    (void)in_sizes; (void)n_in; (void)out_size; (void)ws_size;

    const size_t NX = (size_t)MROWS * D_MODEL;   // 33,554,432 activations
    const size_t NW = (size_t)D_MODEL * D_MODEL; //  4,194,304 per weight

    char*   ws  = (char*)d_ws;
    size_t  off = 0;
    __bf16* xb  = (__bf16*)(ws + off); off += NX * 2;
    __bf16* wqb = (__bf16*)(ws + off); off += NW * 2;
    __bf16* wvb = (__bf16*)(ws + off); off += NW * 2;
    __bf16* wgb = (__bf16*)(ws + off); off += NW * 2;
    __bf16* wob = (__bf16*)(ws + off); off += NW * 2;
    float*  qf  = (float*) (ws + off); off += NX * 4;
    float*  vf  = (float*) (ws + off); off += NX * 4;   // ret computed in place
    float*  gf  = (float*) (ws + off); off += NX * 4;
    __bf16* yb  = (__bf16*)(ws + off); off += NX * 2;   // total ~544 MB

    // 1) fp32 -> bf16 converts
    {
        long nx4 = (long)(NX / 4);
        cvt_f32_bf16<<<(nx4 + 255) / 256, 256, 0, stream>>>(x, xb, nx4);
        long nw4 = (long)(NW / 4);
        cvt_f32_bf16<<<(nw4 + 255) / 256, 256, 0, stream>>>(Wq, wqb, nw4);
        cvt_f32_bf16<<<(nw4 + 255) / 256, 256, 0, stream>>>(Wv, wvb, nw4);
        cvt_f32_bf16<<<(nw4 + 255) / 256, 256, 0, stream>>>(Wg, wgb, nw4);
        cvt_f32_bf16<<<(nw4 + 255) / 256, 256, 0, stream>>>(Wo, wob, nw4);
    }

    // 2) q, v, gate GEMMs (gate with fused SiLU)
    dim3 gg(MROWS / BM, D_MODEL / BN);
    gemm_bf16_wmma<<<gg, 256, 0, stream>>>(xb, wqb, qf, MROWS, D_MODEL, D_MODEL, 0);
    gemm_bf16_wmma<<<gg, 256, 0, stream>>>(xb, wvb, vf, MROWS, D_MODEL, D_MODEL, 0);
    gemm_bf16_wmma<<<gg, 256, 0, stream>>>(xb, wgb, gf, MROWS, D_MODEL, D_MODEL, 1);

    // 3) retention scan (in place on vf)
    retention_scan<<<(BATCH * D_MODEL) / 256, 256, 0, stream>>>(vf, beta);

    // 4) fused LayerNorm + gating -> bf16
    ln_gate_kernel<<<MROWS, 256, 0, stream>>>(qf, vf, gf, gma, btl, yb);

    // 5) output projection -> fp32 d_out
    gemm_bf16_wmma<<<gg, 256, 0, stream>>>(yb, wob, (float*)d_out,
                                           MROWS, D_MODEL, D_MODEL, 0);
}
